// SmoothAP_35485019799839
// MI455X (gfx1250) — compile-verified, matrix operand order checked by reading, not run on hardware
//
#include <hip/hip_runtime.h>

// SmoothAP on gfx1250 (MI455X, wave32).
// D[i,j] = sigmoid((s_j - s_i)*1000) = 0.5*tanh((s_j-s_i)*500) + 0.5
// Per 16x32 tile:  C(16x16) += A([ones; mask; 0...], f16) x B(tanh tile, f16)
//   C row 0 = S0 = sum_k t_k        (lanes 0-15, c[0], N = row-in-tile)
//   C row 1 = S1 = sum_k t_k * m_k  (lanes 0-15, c[1])
// Affine + diagonal folded into the epilogue:
//   R     = 1 + 0.5*(S0 - 1) + 0.5*N
//   R_pos = (1 + 0.5*(S1 - m_i) + 0.5*Msum) * m_i

typedef __attribute__((ext_vector_type(16))) _Float16 v16h;
typedef __attribute__((ext_vector_type(8)))  _Float16 v8h;
typedef __attribute__((ext_vector_type(8)))  float    v8f;

#define N_PTS   1024
#define N_BATCH 32
#define CHUNKS  4          // gridDim.x ; 4 chunks * 8 waves * 2 tiles * 16 rows = 1024
#define LOG2E   1.4426950408889634f

__device__ __forceinline__ float fast_tanh(float z) {
#if __has_builtin(__builtin_amdgcn_tanhf)
    return __builtin_amdgcn_tanhf(z);                 // v_tanh_f32 (1 TRANS op)
#elif __has_builtin(__builtin_amdgcn_tanh_f32)
    return __builtin_amdgcn_tanh_f32(z);
#else
    // tanh(z) = 2*sigmoid(2z) - 1 (exp2 handles saturation via 0/inf -> rcp)
    float e = __builtin_amdgcn_exp2f(-2.0f * z * LOG2E);
    return 2.0f * __builtin_amdgcn_rcpf(1.0f + e) - 1.0f;
#endif
}

__global__ void __launch_bounds__(256)
smoothap_main_kernel(const float* __restrict__ sim,
                     const int*   __restrict__ posmask,   // [B, N+1] (skip col 0)
                     float*       __restrict__ partials)  // [B, CHUNKS]
{
    const int b     = blockIdx.y;
    const int chunk = blockIdx.x;          // 0..3

    __shared__ float    s_s500[N_PTS];     // scores * 500
    __shared__ _Float16 s_mh[N_PTS];       // mask as f16 (0/1)
    __shared__ float    s_red[8];          // per-wave mask counts
    __shared__ float    s_part[8];         // per-wave AP partials

    const float* srow = sim + (size_t)b * N_PTS;
    const int*   mrow = posmask + (size_t)b * (N_PTS + 1) + 1;

    const int lane  = threadIdx.x & 31;
    const int wave  = threadIdx.x >> 5;    // 0..7
    const int colId = lane & 15;
    const int hi    = lane >> 4;

    // ---- stage scores(*500) + mask(f16) into LDS; count mask in-flight ----
    __builtin_prefetch(srow + threadIdx.x, 0, 1);      // global_prefetch_b8
    float cnt = 0.0f;
    for (int idx = threadIdx.x; idx < N_PTS; idx += 256) {
        s_s500[idx] = srow[idx] * 500.0f;
        const float m = (mrow[idx] != 0) ? 1.0f : 0.0f;
        s_mh[idx] = (_Float16)m;
        cnt += m;
    }
    #pragma unroll
    for (int off = 16; off > 0; off >>= 1) cnt += __shfl_down(cnt, off, 32);
    if (lane == 0) s_red[wave] = cnt;
    __syncthreads();

    float msum = 0.0f;
    #pragma unroll
    for (int k = 0; k < 8; ++k) msum += s_red[k];

    // ---- per-wave tiles: rows [iBase0, iBase0+16) and [iBase1, iBase1+16) ----
    const int tile0  = (chunk * 8 + wave) * 2;         // 0..62 step 2
    const int iBase0 = tile0 * 16;
    const int iBase1 = iBase0 + 16;

    // Packed-half t -> K offset (16-bit 16x32 A / 32x16 B layout):
    int koff[16];
    #pragma unroll
    for (int t = 0; t < 16; ++t)
        koff[t] = ((t < 8) ? t : t + 8) + (hi ? 8 : 0);

    // Each lane builds one D-row of B: row iBase + colId, K-half selected by hi.
    const float si0 = s_s500[iBase0 + colId];
    const float si1 = s_s500[iBase1 + colId];

    // A weights: row 0 = ones (lanes 0/16), row 1 = mask slice (lanes 1/17).
    v16h awBase = {};
    if (colId == 0) {
        #pragma unroll
        for (int t = 0; t < 16; ++t) awBase[t] = (_Float16)1.0f;
    }

    v8f c0 = {}, c1 = {};

    for (int step = 0; step < 32; ++step) {
        const int jBase = step * 32;

        // shared s_j loads for both tiles (4x ds_load_b128 pattern)
        float sj[16];
        #pragma unroll
        for (int t = 0; t < 16; ++t) sj[t] = s_s500[jBase + koff[t]];

        // A: refresh mask row (2x ds_load_b128 of f16, lanes 1/17 only)
        v16h aw = awBase;
        if (colId == 1) {
            const v8h mlo = *(const v8h*)&s_mh[jBase + hi * 8];
            const v8h mhi = *(const v8h*)&s_mh[jBase + 16 + hi * 8];
            #pragma unroll
            for (int t = 0; t < 8; ++t) { aw[t] = mlo[t]; aw[t + 8] = mhi[t]; }
        }

        // B tiles: tanh((s_j - s_i)) values, f16
        v16h b0, b1;
        #pragma unroll
        for (int t = 0; t < 16; ++t) {
            b0[t] = (_Float16)fast_tanh(sj[t] - si0);
            b1[t] = (_Float16)fast_tanh(sj[t] - si1);
        }

        c0 = __builtin_amdgcn_wmma_f32_16x16x32_f16(
                 false, aw, false, b0, (short)0, c0, false, false);
        c1 = __builtin_amdgcn_wmma_f32_16x16x32_f16(
                 false, aw, false, b1, (short)0, c1, false, false);
    }

    // ---- epilogue: lanes 0-15 each own one row per tile (c[0]=S0, c[1]=S1) ----
    float part = 0.0f;
    if (lane < 16) {
        #pragma unroll
        for (int r = 0; r < 2; ++r) {
            const int   row = (r ? iBase1 : iBase0) + lane;
            const v8f&  c   = r ? c1 : c0;
            const float mi  = (float)s_mh[row];
            const float S0  = c[0];
            const float S1  = c[1];
            const float R    = 1.0f + 0.5f * (S0 - 1.0f) + 0.5f * (float)N_PTS;
            const float Rpos = (1.0f + 0.5f * (S1 - mi) + 0.5f * msum) * mi;
            part += Rpos * __builtin_amdgcn_rcpf(R);
        }
    }
    #pragma unroll
    for (int off = 16; off > 0; off >>= 1) part += __shfl_down(part, off, 32);
    if (lane == 0) s_part[wave] = part;
    __syncthreads();

    if (threadIdx.x == 0) {
        float acc = 0.0f;
        #pragma unroll
        for (int k = 0; k < 8; ++k) acc += s_part[k];
        partials[b * CHUNKS + chunk] = acc;
    }
}

__global__ void __launch_bounds__(256)
smoothap_finalize_kernel(const int*   __restrict__ posmask,
                         const float* __restrict__ partials,  // [B, CHUNKS]
                         float*       __restrict__ out)       // [B]
{
    const int b = blockIdx.x;
    __shared__ float red[256];

    const int* mrow = posmask + (size_t)b * (N_PTS + 1) + 1;
    float msum = 0.0f;
    for (int idx = threadIdx.x; idx < N_PTS; idx += 256)
        msum += (mrow[idx] != 0) ? 1.0f : 0.0f;
    red[threadIdx.x] = msum;
    __syncthreads();
    for (int s = 128; s > 0; s >>= 1) {
        if (threadIdx.x < s) red[threadIdx.x] += red[threadIdx.x + s];
        __syncthreads();
    }

    if (threadIdx.x == 0) {
        float num = 0.0f;
        #pragma unroll
        for (int k = 0; k < CHUNKS; ++k) num += partials[b * CHUNKS + k];
        out[b] = num / red[0];
    }
}

extern "C" void kernel_launch(void* const* d_in, const int* in_sizes, int n_in,
                              void* d_out, int out_size, void* d_ws, size_t ws_size,
                              hipStream_t stream) {
    const float* sim = (const float*)d_in[0];       // [32, 1024] f32
    const int*   pm  = (const int*)d_in[1];         // [32, 1025] int (bool)
    float*       out = (float*)d_out;               // [32] f32
    float*       ws  = (float*)d_ws;                // needs 32*CHUNKS floats

    dim3 grid1(CHUNKS, N_BATCH);
    smoothap_main_kernel<<<grid1, 256, 0, stream>>>(sim, pm, ws);
    smoothap_finalize_kernel<<<N_BATCH, 256, 0, stream>>>(pm, ws, out);
}
